// AttentionLayer_67070209294585
// MI455X (gfx1250) — matrix operand order repaired and checked by reading, not compiled, not run on hardware
//
#include <hip/hip_runtime.h>
#include <hip/hip_bf16.h>

typedef _Float16 h16;
typedef __attribute__((ext_vector_type(16))) _Float16 v16h;
typedef __attribute__((ext_vector_type(8)))  float    v8f;
typedef __attribute__((ext_vector_type(4)))  unsigned int v4u;
typedef __attribute__((ext_vector_type(8)))  int      v8i;
typedef __attribute__((ext_vector_type(4)))  int      v4i;

union V16 { v16h v; _Float16 e[16]; };

#define D_IN  256
#define D_QK  128
#define SEQ   2048

// ---------------------------------------------------------------------------
// Kernel 1: Q = relu(x*Wq + bq), K = relu(x*Wk + bk), stored f16 in workspace.
// Block = 256 threads = 8 waves; each wave computes a 16-row strip.
// W is staged transposed (f16) in LDS so each WMMA B-operand (32x16, lane=N,
// 16 contiguous K per lane) is a single contiguous 32B LDS read.
// ---------------------------------------------------------------------------
__global__ __launch_bounds__(256) void qk_proj_kernel(
    const float* __restrict__ x,
    const float* __restrict__ Wq, const float* __restrict__ bq,
    const float* __restrict__ Wk, const float* __restrict__ bk,
    h16* __restrict__ Qh, h16* __restrict__ Kh)
{
    __shared__ __align__(32) h16 WT[2][D_QK * D_IN];  // [proj][f][d] (transposed)

    const int tid = threadIdx.x;

    // Stage both weight matrices, transposed, f32 -> f16.
    for (int p = 0; p < 2; ++p) {
        const float* W = p ? Wk : Wq;
        for (int i = 0; i < (D_IN * D_QK) / 256; ++i) {
            int idx = i * 256 + tid;        // idx = d*128 + f (row-major [256][128])
            int d = idx >> 7, f = idx & 127;
            WT[p][f * D_IN + d] = (h16)W[idx];
        }
    }
    __syncthreads();

    const int wave = tid >> 5, lane = tid & 31;
    const int m  = lane & 15;     // A: row within 16-row strip / C: column
    const int hh = lane >> 4;     // half-wave select (K lo/hi in A, M+8 in C)
    const int rowbase = blockIdx.x * 128 + wave * 16;
    const float* xrow = x + (size_t)(rowbase + m) * D_IN;

    // A operands: 8 chunks of K=32. K order used here (B matches):
    // lane<16 holds K = kc*32+0..15; lane>=16 holds kc*32+16..31.
    V16 a[8];
    for (int kc = 0; kc < 8; ++kc) {
        int b0 = kc * 32 + hh * 16;
        for (int j = 0; j < 8; ++j) {
            a[kc].e[j]     = (h16)xrow[b0 + j];
            a[kc].e[j + 8] = (h16)xrow[b0 + 8 + j];
        }
    }

    for (int p = 0; p < 2; ++p) {
        const float* bias = p ? bk : bq;
        h16* outp = p ? Kh : Qh;
        for (int nt = 0; nt < 8; ++nt) {
            const int n = nt * 16 + m;
            v8f acc = {};
            for (int kc = 0; kc < 8; ++kc) {
                v16h bmat = *(const v16h*)&WT[p][n * D_IN + kc * 32 + hh * 16];
                acc = __builtin_amdgcn_wmma_f32_16x16x32_f16(
                    false, a[kc].v, false, bmat, (short)0, acc, false, false);
            }
            float bv = bias[n];
            // C layout: VGPR r, lane -> (M = r + 8*hh, N = lane&15)
            for (int r = 0; r < 8; ++r) {
                float v = fmaxf(acc[r] + bv, 0.0f);
                outp[(size_t)(rowbase + r + 8 * hh) * D_QK + n] = (h16)v;
            }
        }
    }
}

// ---------------------------------------------------------------------------
// Kernel 2: flash attention over the original inputs as V.
// Block = (q-tile of 128 rows) x batch; 8 waves, each owns 16 query rows and a
// full 16x256 f32 accumulator in VGPRs. Per 128-key tile:
//   * K-tile (32KB f16, contiguous) DMA'd to LDS by the Tensor Data Mover
//   * V staged transposed f16 in LDS (needs f32->f16 convert, so VALU path)
//   * scores via WMMA, online softmax with shfl reductions, P bounced through
//     wave-private LDS to re-layout C->A, then P@V via WMMA.
// ---------------------------------------------------------------------------
__global__ __launch_bounds__(256) void attn_kernel(
    const float* __restrict__ x,
    const h16* __restrict__ Qh, const h16* __restrict__ Kh,
    float* __restrict__ out)
{
    __shared__ __align__(32) h16 ldsK[128 * D_QK];   // [key][128]  (TDM target)
    __shared__ __align__(32) h16 Vt[D_IN * 128];     // [dim][key]  (transposed V)
    __shared__ __align__(32) h16 Pt[8 * 16 * 128];   // [wave][m][key]

    const int tid  = threadIdx.x;
    const int wave = tid >> 5, lane = tid & 31;
    const int m  = lane & 15;
    const int hh = lane >> 4;
    const int qt = blockIdx.x, b = blockIdx.y;
    const long qrow0 = (long)b * SEQ + qt * 128 + wave * 16;

    // Q strip A-operands: 4 chunks of K=32, contiguous f16 loads from global.
    V16 aq[4];
    {
        const h16* qrow = Qh + (size_t)(qrow0 + m) * D_QK;
        for (int kc = 0; kc < 4; ++kc) {
            int b0 = kc * 32 + hh * 16;
            for (int j = 0; j < 8; ++j) {
                aq[kc].e[j]     = qrow[b0 + j];
                aq[kc].e[j + 8] = qrow[b0 + 8 + j];
            }
        }
    }

    float mrun[8], lrun[8];
    v8f O[16];
    for (int r = 0; r < 8; ++r) { mrun[r] = -__builtin_inff(); lrun[r] = 0.0f; }
    for (int nt = 0; nt < 16; ++nt) O[nt] = (v8f){};

    h16* prow = &Pt[wave * 16 * 128];

    for (int kt = 0; kt < 16; ++kt) {
        const long krow0 = (long)b * SEQ + kt * 128;
        __syncthreads();   // previous tile fully consumed before restaging

        if (tid < 32) {    // wave 0 issues the tensor DMA (EXEC ignored by TDM)
            unsigned lds_off = (unsigned)(unsigned long long)(const void*)&ldsK[0];
            unsigned long long ga =
                (unsigned long long)(const void*)(Kh + (size_t)krow0 * D_QK);
            // D# group0: count=1 | lds_addr | global_addr | type=2
            v4u g0 = { 1u, lds_off,
                       (unsigned)(ga & 0xffffffffu),
                       (unsigned)((ga >> 32) & 0x1ffffffu) | (2u << 30) };
            // D# group1: data_size=2B; tensor_dim0=0x200000 elems (whole Kh);
            // tensor_dim1=1; tile_dim0=16384 elems (=32KB contiguous); tile_dim1=1;
            // tensor_dim0_stride=0x200000.
            v8i g1 = { (int)(1u << 16),
                       0,
                       (int)0x00010020,
                       (int)0x40000000,
                       1,
                       (int)0x00200000,
                       0, 0 };
            v4i g2 = {0, 0, 0, 0};
            v4i g3 = {0, 0, 0, 0};
            v8i g4 = {0, 0, 0, 0, 0, 0, 0, 0};
            __builtin_amdgcn_tensor_load_to_lds(g0, g1, g2, g3, g4, 0);
            __builtin_amdgcn_s_wait_tensorcnt(0);
        }

        // Stage V transposed: thread 'tid' owns dim=tid, walks 128 keys.
        for (int i = 0; i < 128; ++i)
            Vt[tid * 128 + i] = (h16)x[(size_t)(krow0 + i) * D_IN + tid];
        __syncthreads();

        // ---- scores: S[16 q-rows][128 keys] for this wave ----
        v8f S[8];
        for (int kn = 0; kn < 8; ++kn) {
            v8f acc = {};
            const h16* kr = &ldsK[(kn * 16 + m) * D_QK];   // lane = key column
            for (int kc = 0; kc < 4; ++kc) {
                v16h bk_ = *(const v16h*)&kr[kc * 32 + hh * 16];
                acc = __builtin_amdgcn_wmma_f32_16x16x32_f16(
                    false, aq[kc].v, false, bk_, (short)0, acc, false, false);
            }
            S[kn] = acc;
        }

        // ---- online softmax (row stats live per (r, half-wave)) ----
        float mnew[8], psum[8];
        for (int r = 0; r < 8; ++r) {
            float t = S[0][r];
            for (int kn = 1; kn < 8; ++kn) t = fmaxf(t, S[kn][r]);
            t = fmaxf(t, __shfl_xor(t, 1, 32));
            t = fmaxf(t, __shfl_xor(t, 2, 32));
            t = fmaxf(t, __shfl_xor(t, 4, 32));
            t = fmaxf(t, __shfl_xor(t, 8, 32));
            mnew[r] = fmaxf(mrun[r], t);
            float scale = __expf(mrun[r] - mnew[r]);
            lrun[r] *= scale;
            mrun[r]  = mnew[r];
            for (int nt = 0; nt < 16; ++nt) O[nt][r] *= scale;
            psum[r] = 0.0f;
        }
        // exp + write P (f16) transposed into wave-private LDS (C -> A relayout)
        for (int kn = 0; kn < 8; ++kn) {
            for (int r = 0; r < 8; ++r) {
                float p = __expf(S[kn][r] - mnew[r]);
                psum[r] += p;
                prow[(r + 8 * hh) * 128 + kn * 16 + m] = (h16)p;
            }
        }
        for (int r = 0; r < 8; ++r) {
            float t = psum[r];
            t += __shfl_xor(t, 1, 32);
            t += __shfl_xor(t, 2, 32);
            t += __shfl_xor(t, 4, 32);
            t += __shfl_xor(t, 8, 32);
            lrun[r] += t;
        }
        asm volatile("s_wait_dscnt 0" ::: "memory");  // P visible across lanes

        // ---- O += P @ V ----
        for (int kc = 0; kc < 4; ++kc) {
            V16 ap;
            const h16* pr = &prow[m * 128 + kc * 32];
            for (int j = 0; j < 8; ++j) {
                ap.e[j]     = pr[hh * 16 + j];
                ap.e[j + 8] = pr[hh * 16 + 8 + j];
            }
            for (int nt = 0; nt < 16; ++nt) {
                v16h bv = *(const v16h*)&Vt[(nt * 16 + m) * 128 + kc * 32 + hh * 16];
                O[nt] = __builtin_amdgcn_wmma_f32_16x16x32_f16(
                    false, ap.v, false, bv, (short)0, O[nt], false, false);
            }
        }
    }

    // ---- normalize and store ----
    for (int r = 0; r < 8; ++r) {
        float inv = 1.0f / lrun[r];
        long row = qrow0 + r + 8 * hh;
        for (int nt = 0; nt < 16; ++nt)
            out[row * D_IN + nt * 16 + m] = O[nt][r] * inv;
    }
}

// ---------------------------------------------------------------------------
extern "C" void kernel_launch(void* const* d_in, const int* in_sizes, int n_in,
                              void* d_out, int out_size, void* d_ws, size_t ws_size,
                              hipStream_t stream)
{
    const float* x  = (const float*)d_in[0];
    const float* Wq = (const float*)d_in[1];
    const float* bq = (const float*)d_in[2];
    const float* Wk = (const float*)d_in[3];
    const float* bk = (const float*)d_in[4];
    float* out = (float*)d_out;

    const int nRows = in_sizes[0] / D_IN;      // B*S = 16384
    const int B = nRows / SEQ;

    h16* Qh = (h16*)d_ws;                       // nRows*128 f16
    h16* Kh = Qh + (size_t)nRows * D_QK;        // nRows*128 f16

    qk_proj_kernel<<<nRows / 128, 256, 0, stream>>>(x, Wq, bq, Wk, bk, Qh, Kh);
    attn_kernel<<<dim3(SEQ / 128, B), 256, 0, stream>>>(x, Qh, Kh, out);
}
